// LatticeMultiHeadAttention_16733192585388
// MI455X (gfx1250) — compile-verified, matrix operand order checked by reading, not run on hardware
//
#include <hip/hip_runtime.h>
#include <hip/hip_bf16.h>
#include <math.h>

#define D_MODEL 1024
#define N_HEADS 16
#define D_K     64
#define BATCH   8
#define SEQ     1024
#define BL      (BATCH * SEQ)          // 8192 rows
#define TWO_PI_F 6.2831853071795864f

#if defined(__has_builtin)
#  if __has_builtin(__builtin_amdgcn_global_load_async_to_lds_b128)
#    define USE_ASYNC_LDS 1
#  endif
#endif
#ifndef USE_ASYNC_LDS
#  define USE_ASYNC_LDS 0
#endif

typedef __attribute__((ext_vector_type(16))) __bf16 v16bf;
typedef __attribute__((ext_vector_type(8)))  float  v8f;
typedef __attribute__((ext_vector_type(4)))  int    i32x4;

typedef __attribute__((address_space(1))) i32x4 glb_i32x4;
typedef __attribute__((address_space(3))) i32x4 lds_i32x4;

union Frag16 {
    v16bf v;
    unsigned int u[8];
    uint4 q[2];          // q[0] covers K-pairs v0..3, q[1] covers v4..7
};

// round-to-nearest-even fp32 -> bf16 bit pattern
__device__ __forceinline__ unsigned short f2bf(float f) {
    unsigned int u = __float_as_uint(f);
    u += 0x7FFFu + ((u >> 16) & 1u);
    return (unsigned short)(u >> 16);
}

// Fragment K-pair striping (ISA 7.12.2 16x32 A layout): v=0..3 at hf*8+2v,
// v=4..7 at 16+hf*8+2(v&3) -> two contiguous 16B groups per fragment.
__device__ __forceinline__ void load_frag_contig(Frag16& f,
                                                 const unsigned short* base,
                                                 int hf) {
    f.q[0] = *(const uint4*)(base + hf * 8);
    f.q[1] = *(const uint4*)(base + 16 + hf * 8);
}

__device__ __forceinline__ void async_copy16(const unsigned short* gsrc,
                                             unsigned short* ldst) {
#if USE_ASYNC_LDS
    __builtin_amdgcn_global_load_async_to_lds_b128(
        (glb_i32x4*)gsrc, (lds_i32x4*)ldst, 0, 0);
#else
    *(uint4*)ldst = *(const uint4*)gsrc;
#endif
}

__device__ __forceinline__ void wait_async0() {
#if USE_ASYNC_LDS
#  if __has_builtin(__builtin_amdgcn_s_wait_asynccnt)
    __builtin_amdgcn_s_wait_asynccnt(0);
#  else
    asm volatile("s_wait_asynccnt 0x0" ::: "memory");
#  endif
#endif
}

// ---------------------------------------------------------------------------
// Kernel 1: Kuramoto phase update -> per-head additive bias (16 heads)
// ---------------------------------------------------------------------------
__global__ void lattice_bias_kernel(const float* __restrict__ phases0,
                                    const float* __restrict__ freqs,
                                    const float* __restrict__ nbr_w,
                                    const int*   __restrict__ nbr_idx,
                                    float* __restrict__ bias) {
    int h = threadIdx.x;
    if (h >= N_HEADS) return;
    float ph = phases0[h];
    float force = 0.0f;
    #pragma unroll
    for (int k = 0; k < 8; ++k) {
        int   idx   = nbr_idx[h * 8 + k];
        float valid = (idx >= 0) ? 1.0f : 0.0f;
        int   ci    = (idx < 0) ? 0 : idx;
        float nph   = phases0[ci] * valid;
        force += nbr_w[h * 8 + k] * sinf(nph - ph);
    }
    float f  = freqs[h];
    float p1 = fmodf(ph + 0.01f * (f + force), TWO_PI_F);
    if (p1 < 0.0f) p1 += TWO_PI_F;              // jnp.mod is floored
    float p2 = fmodf(p1 + 0.01f * f, TWO_PI_F);
    if (p2 < 0.0f) p2 += TWO_PI_F;
    bias[h] = 0.1f * cosf(p2);
}

// ---------------------------------------------------------------------------
// Kernel 2a: fp32 -> bf16 bulk convert (activations)
// 8 elements per thread, b128 in / b128 out.
// ---------------------------------------------------------------------------
__global__ __launch_bounds__(256)
void convert_x_kernel(const float* __restrict__ src,
                      unsigned short* __restrict__ dst) {
    size_t i = ((size_t)blockIdx.x * 256 + threadIdx.x) * 8;
    const float4* sp = (const float4*)(src + i);
    float4 a = sp[0], b = sp[1];
    float f[8] = {a.x, a.y, a.z, a.w, b.x, b.y, b.z, b.w};
    uint4 r;
    r.x = f2bf(f[0]) | ((unsigned)f2bf(f[1]) << 16);
    r.y = f2bf(f[2]) | ((unsigned)f2bf(f[3]) << 16);
    r.z = f2bf(f[4]) | ((unsigned)f2bf(f[5]) << 16);
    r.w = f2bf(f[6]) | ((unsigned)f2bf(f[7]) << 16);
    *(uint4*)(dst + i) = r;
}

// ---------------------------------------------------------------------------
// Kernel 2b: weight convert -> bf16, pre-transposed to Wt[col][k] so GEMM B
// tiles are contiguous along K.
//   headStacked=1: W is [H, D_MODEL, D_K], col = h*64 + kk
//   headStacked=0: W is [D_MODEL, D_MODEL] (Wo), Wt[col][k] = W[k][col]
// ---------------------------------------------------------------------------
__global__ __launch_bounds__(256)
void convert_w_kernel(const float* __restrict__ W,
                      unsigned short* __restrict__ Wt,
                      int headStacked) {
    int id  = blockIdx.x * 256 + threadIdx.x;    // 1024 cols * 128 k-chunks
    int col = id >> 7;
    int k8  = (id & 127) * 8;
    int h   = col >> 6, kk = col & 63;
    unsigned int u[4];
    #pragma unroll
    for (int e = 0; e < 4; ++e) {
        float f0, f1;
        if (headStacked) {
            f0 = W[((size_t)h * D_MODEL + k8 + 2 * e)     * D_K + kk];
            f1 = W[((size_t)h * D_MODEL + k8 + 2 * e + 1) * D_K + kk];
        } else {
            f0 = W[(size_t)(k8 + 2 * e)     * D_MODEL + col];
            f1 = W[(size_t)(k8 + 2 * e + 1) * D_MODEL + col];
        }
        u[e] = f2bf(f0) | ((unsigned)f2bf(f1) << 16);
    }
    uint4 r; r.x = u[0]; r.y = u[1]; r.z = u[2]; r.w = u[3];
    *(uint4*)(Wt + (size_t)col * D_MODEL + k8) = r;
}

// ---------------------------------------------------------------------------
// Kernel 3: bf16 WMMA GEMM, C[M,1024] = Abf[M,1024] @ Wt^T
//   Abf bf16 row-major, Wt bf16 [col][k] (pre-transposed).
//   mode 0: out bf16 -> [B,H,L,64]   (Q, K)
//   mode 2: out bf16 -> [B,H,64,L]   (V transposed)
//   mode 1: out fp32 + bo -> outF[M,1024]   (out-proj)
// Block: 256 threads (8 waves, 2x4), 64x128 tile, 2x2 WMMA tiles per wave.
// Tiles copied global->LDS with GLOBAL_LOAD_ASYNC_TO_LDS_B128 (ASYNCcnt DMA).
// ---------------------------------------------------------------------------
#define BM 64
#define BN 128
#define BKT 32
#define LDS_STRIDE 40    // ushorts; 80 B row stride, multiple of 16 B

__global__ __launch_bounds__(256)
void wmma_gemm_kernel(const unsigned short* __restrict__ Abf,
                      const unsigned short* __restrict__ Wt,
                      const float* __restrict__ bo,
                      unsigned short* __restrict__ outBf,
                      float* __restrict__ outF,
                      int mode) {
    __shared__ unsigned short Al[BM][LDS_STRIDE];    // [m][k]
    __shared__ unsigned short Blt[BN][LDS_STRIDE];   // [n][k]

    const int tid  = threadIdx.x;
    const int lane = tid & 31;
    const int wave = tid >> 5;
    const int wm   = wave >> 2;          // 0..1  (32-row slabs)
    const int wn   = wave & 3;           // 0..3  (32-col slabs)
    const int row0 = blockIdx.y * BM;
    const int col0 = blockIdx.x * BN;
    const int n    = lane & 15;
    const int hf   = lane >> 4;

    v8f c[2][2];
    c[0][0] = 0; c[0][1] = 0; c[1][0] = 0; c[1][1] = 0;

    // A: 64 rows x 32 k = 256 16B chunks, one per thread
    const int rowA = tid >> 2, chA = (tid & 3) * 8;
    // B: 128 rows x 32 k = 512 16B chunks, two per thread
    const int rowB = tid >> 1, chB = (tid & 1) * 16;

    for (int k0 = 0; k0 < D_MODEL; k0 += BKT) {
        const unsigned short* gA = Abf + (size_t)(row0 + rowA) * D_MODEL + k0 + chA;
        const unsigned short* gB = Wt  + (size_t)(col0 + rowB) * D_MODEL + k0 + chB;
        async_copy16(gA,     &Al[rowA][chA]);
        async_copy16(gB,     &Blt[rowB][chB]);
        async_copy16(gB + 8, &Blt[rowB][chB + 8]);
        wait_async0();
        __syncthreads();

        // fragments: each is two ds_load_b128
        Frag16 a[2], b[2];
        #pragma unroll
        for (int i = 0; i < 2; ++i)
            load_frag_contig(a[i], &Al[wm * 32 + i * 16 + n][0], hf);
        #pragma unroll
        for (int j = 0; j < 2; ++j)
            load_frag_contig(b[j], &Blt[wn * 32 + j * 16 + n][0], hf);

        #pragma unroll
        for (int i = 0; i < 2; ++i)
            #pragma unroll
            for (int j = 0; j < 2; ++j)
                c[i][j] = __builtin_amdgcn_wmma_f32_16x16x32_bf16(
                    false, a[i].v, false, b[j].v, (short)0, c[i][j], false, false);
        __syncthreads();
    }

    // epilogue: C fragment lane -> N, VGPR v -> row v + 8*hf
    #pragma unroll
    for (int i = 0; i < 2; ++i) {
        const int rbase = row0 + wm * 32 + i * 16 + hf * 8;
        #pragma unroll
        for (int j = 0; j < 2; ++j) {
            const int cgl = col0 + wn * 32 + j * 16 + n;
            if (mode == 1) {
                const float bias = bo[cgl];
                #pragma unroll
                for (int v = 0; v < 8; ++v)
                    outF[(size_t)(rbase + v) * D_MODEL + cgl] = c[i][j][v] + bias;
            } else {
                const int h = cgl >> 6, kk = cgl & 63;
                #pragma unroll
                for (int v = 0; v < 8; ++v) {
                    int row = rbase + v;                // flat row in [0, B*L)
                    int bb = row >> 10, l = row & 1023;
                    size_t headBase = ((size_t)bb * N_HEADS + h);
                    if (mode == 0)
                        outBf[(headBase * SEQ + l) * D_K + kk] = f2bf(c[i][j][v]);
                    else                                 // mode 2: [B,H,D_K,SEQ]
                        outBf[(headBase * D_K + kk) * SEQ + l] = f2bf(c[i][j][v]);
                }
            }
        }
    }
}

// ---------------------------------------------------------------------------
// Kernel 4: flash attention, one wave per 16 query rows per (b,h).
// Q,K bf16 [B,H,L,64]; Vt bf16 [B,H,64,L]; ctx bf16 [B,L,H*64]
// ---------------------------------------------------------------------------
__global__ __launch_bounds__(32)
void flash_attn_kernel(const unsigned short* __restrict__ Q,
                       const unsigned short* __restrict__ K,
                       const unsigned short* __restrict__ Vt,
                       const float* __restrict__ bias,
                       unsigned short* __restrict__ ctx) {
    __shared__ unsigned short Pl[16][LDS_STRIDE];    // P tile 16x32

    const int lane = threadIdx.x;
    const int n    = lane & 15;
    const int hf   = lane >> 4;
    const int q0   = blockIdx.x * 16;
    const int h    = blockIdx.y;
    const int b    = blockIdx.z;

    const size_t head = ((size_t)b * N_HEADS + h);
    const unsigned short* Qh  = Q  + head * SEQ * D_K;
    const unsigned short* Kh  = K  + head * SEQ * D_K;
    const unsigned short* Vth = Vt + head * D_K * SEQ;
    const float bh    = bias[h];
    const float scale = 0.125f;                      // 1/sqrt(64)

    // Q fragments for the two 32-wide k-steps over d_k (two b128 loads each)
    Frag16 aq[2];
    #pragma unroll
    for (int ks = 0; ks < 2; ++ks)
        load_frag_contig(aq[ks], Qh + (size_t)(q0 + n) * D_K + ks * 32, hf);

    float mrow[8], lrow[8];
    v8f acc[4];
    #pragma unroll
    for (int v = 0; v < 8; ++v) { mrow[v] = -1e30f; lrow[v] = 0.0f; }
    #pragma unroll
    for (int t = 0; t < 4; ++t) acc[t] = 0;

    for (int kt = 0; kt < SEQ; kt += 32) {
        // ---- scores S[16 x 32] = Q @ K^T (two 16-key sub-tiles) ----
        v8f s[2]; s[0] = 0; s[1] = 0;
        #pragma unroll
        for (int sub = 0; sub < 2; ++sub) {
            #pragma unroll
            for (int ks = 0; ks < 2; ++ks) {
                Frag16 bk;   // B[d, key] = K[key, d]: K-pairs contiguous in d
                load_frag_contig(
                    bk, Kh + (size_t)(kt + sub * 16 + n) * D_K + ks * 32, hf);
                s[sub] = __builtin_amdgcn_wmma_f32_16x16x32_bf16(
                    false, aq[ks].v, false, bk.v, (short)0, s[sub], false, false);
            }
        }

        // ---- online softmax (reductions within each 16-lane half) ----
        float alpha[8];
        #pragma unroll
        for (int v = 0; v < 8; ++v) {
            s[0][v] = s[0][v] * scale + bh;
            s[1][v] = s[1][v] * scale + bh;
            float t = fmaxf(s[0][v], s[1][v]);
            #pragma unroll
            for (int mask = 1; mask < 16; mask <<= 1)
                t = fmaxf(t, __shfl_xor(t, mask, 32));
            float mnew = fmaxf(mrow[v], t);
            alpha[v]   = __expf(mrow[v] - mnew);
            float p0 = __expf(s[0][v] - mnew);
            float p1 = __expf(s[1][v] - mnew);
            s[0][v] = p0; s[1][v] = p1;
            float r = p0 + p1;
            #pragma unroll
            for (int mask = 1; mask < 16; mask <<= 1)
                r += __shfl_xor(r, mask, 32);
            lrow[v] = lrow[v] * alpha[v] + r;
            mrow[v] = mnew;
        }

        // ---- P (C-layout) -> A-fragment via LDS transpose ----
        __syncthreads();                             // prior reads done
        #pragma unroll
        for (int v = 0; v < 8; ++v) {
            int r = v + hf * 8;                      // score row
            Pl[r][n]      = f2bf(s[0][v]);
            Pl[r][16 + n] = f2bf(s[1][v]);
        }
        #pragma unroll
        for (int t = 0; t < 4; ++t)
            #pragma unroll
            for (int v = 0; v < 8; ++v)
                acc[t][v] *= alpha[v];
        __syncthreads();                             // writes visible

        Frag16 pa;
        load_frag_contig(pa, &Pl[n][0], hf);

        // ---- acc += P @ V over 4 column tiles of d_k ----
        // Vt layout [dk][key]: K-pairs contiguous -> two b128 loads per frag
        #pragma unroll
        for (int t = 0; t < 4; ++t) {
            Frag16 bv;
            load_frag_contig(bv, Vth + (size_t)(t * 16 + n) * SEQ + kt, hf);
            acc[t] = __builtin_amdgcn_wmma_f32_16x16x32_bf16(
                false, pa.v, false, bv.v, (short)0, acc[t], false, false);
        }
    }

    // ---- normalize, write ctx bf16 as [B, L, H*64] ----
    #pragma unroll
    for (int t = 0; t < 4; ++t)
        #pragma unroll
        for (int v = 0; v < 8; ++v) {
            int q = q0 + v + hf * 8;
            ctx[((size_t)b * SEQ + q) * D_MODEL + h * D_K + t * 16 + n] =
                f2bf(acc[t][v] / lrow[v]);
        }
}

// ---------------------------------------------------------------------------
// Launch
// ---------------------------------------------------------------------------
extern "C" void kernel_launch(void* const* d_in, const int* in_sizes, int n_in,
                              void* d_out, int out_size, void* d_ws, size_t ws_size,
                              hipStream_t stream) {
    const float* query   = (const float*)d_in[0];
    const float* key_    = (const float*)d_in[1];
    const float* value   = (const float*)d_in[2];
    const float* Wq      = (const float*)d_in[3];
    const float* Wk      = (const float*)d_in[4];
    const float* Wv      = (const float*)d_in[5];
    const float* Wo      = (const float*)d_in[6];
    const float* bo      = (const float*)d_in[7];
    const float* phases0 = (const float*)d_in[8];
    const float* freqs   = (const float*)d_in[9];
    const float* nbr_w   = (const float*)d_in[10];
    const int*   nbr_idx = (const int*)d_in[11];

    char* ws = (char*)d_ws;
    const size_t xElems = (size_t)BL * D_MODEL;        // 8,388,608
    const size_t wElems = (size_t)D_MODEL * D_MODEL;   // 1,048,576

    float*          biasb = (float*)ws;
    unsigned short* Xq    = (unsigned short*)(ws + 256);
    unsigned short* Xk    = Xq  + xElems;
    unsigned short* Xv    = Xk  + xElems;
    unsigned short* Qws   = Xv  + xElems;
    unsigned short* Kws   = Qws + xElems;
    unsigned short* Vtws  = Kws + xElems;
    unsigned short* ctxBf = Vtws + xElems;
    unsigned short* Wqt   = ctxBf + xElems;
    unsigned short* Wkt   = Wqt + wElems;
    unsigned short* Wvt   = Wkt + wElems;
    unsigned short* Wot   = Wvt + wElems;

    lattice_bias_kernel<<<1, 32, 0, stream>>>(phases0, freqs, nbr_w, nbr_idx, biasb);

    const int xBlocks = (int)(xElems / (256 * 8));     // 4096
    convert_x_kernel<<<xBlocks, 256, 0, stream>>>(query, Xq);
    convert_x_kernel<<<xBlocks, 256, 0, stream>>>(key_, Xk);
    convert_x_kernel<<<xBlocks, 256, 0, stream>>>(value, Xv);

    const int wBlocks = (int)(wElems / (256 * 8));     // 512
    convert_w_kernel<<<wBlocks, 256, 0, stream>>>(Wq, Wqt, 1);
    convert_w_kernel<<<wBlocks, 256, 0, stream>>>(Wk, Wkt, 1);
    convert_w_kernel<<<wBlocks, 256, 0, stream>>>(Wv, Wvt, 1);
    convert_w_kernel<<<wBlocks, 256, 0, stream>>>(Wo, Wot, 0);

    dim3 gemmGrid(D_MODEL / BN, BL / BM);   // (8, 128)
    wmma_gemm_kernel<<<gemmGrid, 256, 0, stream>>>(Xq, Wqt, nullptr, Qws,  nullptr, 0);
    wmma_gemm_kernel<<<gemmGrid, 256, 0, stream>>>(Xk, Wkt, nullptr, Kws,  nullptr, 0);
    wmma_gemm_kernel<<<gemmGrid, 256, 0, stream>>>(Xv, Wvt, nullptr, Vtws, nullptr, 2);

    flash_attn_kernel<<<dim3(SEQ / 16, N_HEADS, BATCH), 32, 0, stream>>>(
        Qws, Kws, Vtws, biasb, ctxBf);

    wmma_gemm_kernel<<<gemmGrid, 256, 0, stream>>>(ctxBf, Wot, bo, nullptr,
                                                   (float*)d_out, 1);
}
